// GraphLearnerSparse_2894807957548
// MI455X (gfx1250) — compile-verified, matrix operand order checked by reading, not run on hardware
//
#include <hip/hip_runtime.h>
#include <hip/hip_bf16.h>

#define NN      40000
#define FDIM    64
#define KNN     24
#define EDG     (NN * KNN)        // 960000
#define EADJ    (NN * KNN)        // 960000
#define OMEGA_C 0.9f

typedef __attribute__((ext_vector_type(16))) __bf16 v16bf;
typedef __attribute__((ext_vector_type(8)))  float  v8f;

// ---------- helpers ----------
__device__ __forceinline__ unsigned short f2bf_rne(float f) {
    unsigned int u = __float_as_uint(f);
    unsigned int r = 0x7FFFu + ((u >> 16) & 1u);
    return (unsigned short)((u + r) >> 16);
}

__device__ __forceinline__ void atomAddF(float* p, float v) {
    unsafeAtomicAdd(p, v);   // lowers to global_atomic_add_f32 (no return)
}

// ---------- conversion kernels ----------
__global__ void k_f2bf(const float* __restrict__ src, unsigned short* __restrict__ dst, int n) {
    int t = blockIdx.x * blockDim.x + threadIdx.x;
    if (t < n) dst[t] = f2bf_rne(src[t]);
}

__global__ void k_relu_f2bf(const float* __restrict__ src, unsigned short* __restrict__ dst, int n) {
    int t = blockIdx.x * blockDim.x + threadIdx.x;
    if (t < n) dst[t] = f2bf_rne(fmaxf(src[t], 0.0f));
}

// transpose + convert W (64x64): Wt[n*64+k] = bf16(W[k*64+n])
// makes the WMMA B-fragment pairs contiguous (same shape as A fragment)
__global__ void k_w_t_bf(const float* __restrict__ W, unsigned short* __restrict__ Wt) {
    int t = blockIdx.x * blockDim.x + threadIdx.x;   // 4096 threads
    int n = t >> 6, k = t & 63;
    Wt[t] = f2bf_rne(W[k * FDIM + n]);
}

// ---------- degree / dinv (GCN, with self loop) ----------
__global__ void k_fill(float* __restrict__ p, float v, int n) {
    int t = blockIdx.x * blockDim.x + threadIdx.x;
    if (t < n) p[t] = v;
}

__global__ void k_deg_add(const int* __restrict__ col, float* __restrict__ deg, int n) {
    int t = blockIdx.x * blockDim.x + threadIdx.x;
    if (t < n) atomAddF(&deg[col[t]], 1.0f);
}

__global__ void k_rsqrt(const float* __restrict__ deg, float* __restrict__ dinv, int n) {
    int t = blockIdx.x * blockDim.x + threadIdx.x;
    if (t < n) dinv[t] = rsqrtf(deg[t]);   // deg >= 1 always (self loop)
}

// ---------- WMMA GEMM: C[NN,64] = A[NN,64](bf16) x W[64,64] (Bt = W^T, bf16) ----------
// One wave per 16x16 C tile. 2500 M-tiles x 4 N-tiles = 10000 waves.
// A fragment: ISA 16-bit 16x32 layout (lanes 0-15: M=lane, K pairs 0..7/16..23;
// lanes 16-31: M=lane-16, K pairs 8..15/24..31). B fragment mirrors with N=lane%16,
// reading Bt[n][k] = W[k][n] so pairs are contiguous -> b128 loads.
__global__ void k_gemm_bf16(const unsigned short* __restrict__ A,
                            const unsigned short* __restrict__ Bt,
                            float* __restrict__ C) {
    int wave  = (blockIdx.x * blockDim.x + threadIdx.x) >> 5;
    int lane  = threadIdx.x & 31;
    int mtile = wave >> 2;
    int ntile = wave & 3;
    int half  = lane >> 4;
    int l16   = lane & 15;
    int m     = mtile * 16 + l16;      // A row held by this lane
    int n     = ntile * 16 + l16;      // B/C column held by this lane

    v8f c = {};
    #pragma unroll
    for (int s = 0; s < 2; ++s) {      // K = 64 -> two K=32 steps
        union { unsigned short u[16]; v16bf v; } a, b;
        #pragma unroll
        for (int r = 0; r < 8; ++r) {
            int kb = ((r < 4) ? 2 * r : 16 + 2 * (r - 4)) + 8 * half + 32 * s;
            a.u[2 * r]     = A[m * FDIM + kb];
            a.u[2 * r + 1] = A[m * FDIM + kb + 1];
            b.u[2 * r]     = Bt[n * FDIM + kb];
            b.u[2 * r + 1] = Bt[n * FDIM + kb + 1];
        }
        c = __builtin_amdgcn_wmma_f32_16x16x32_bf16(false, a.v, false, b.v,
                                                    (short)0, c, false, false);
    }
    #pragma unroll
    for (int r = 0; r < 8; ++r) {
        int mr = mtile * 16 + r + 8 * half;
        C[mr * FDIM + n] = c[r];
    }
}

// ---------- GCN epilogue: h = dinv^2 * xw + bias (self loop), then aggregate ----------
__global__ void k_selfloop_init(const float* __restrict__ xw, const float* __restrict__ dinv,
                                const float* __restrict__ bias, float* __restrict__ h) {
    int t = blockIdx.x * blockDim.x + threadIdx.x;
    if (t >= NN * FDIM) return;
    int m = t >> 6, f = t & 63;
    float d = dinv[m];
    h[t] = d * d * xw[t] + bias[f];
}

__global__ void k_aggregate(const float* __restrict__ xw,
                            const int* __restrict__ row, const int* __restrict__ col,
                            const float* __restrict__ dinv, float* __restrict__ h) {
    int t = blockIdx.x * blockDim.x + threadIdx.x;   // EADJ*64 threads
    int e = t >> 6, f = t & 63;
    int r = row[e], c = col[e];
    float v = dinv[r] * dinv[c] * xw[r * FDIM + f];
    atomAddF(&h[c * FDIM + f], v);
}

// ---------- row L2-normalize: one wave per node ----------
__global__ void k_normalize(float* __restrict__ h) {
    int wave = (blockIdx.x * blockDim.x + threadIdx.x) >> 5;
    int lane = threadIdx.x & 31;
    const float2* p = (const float2*)&h[wave * FDIM + lane * 2];
    float2 a = *p;
    float ss = a.x * a.x + a.y * a.y;
    #pragma unroll
    for (int off = 16; off > 0; off >>= 1) ss += __shfl_xor(ss, off, 32);
    float scale = 1.0f / fmaxf(sqrtf(ss), 1e-12f);
    float2 o; o.x = a.x * scale; o.y = a.y * scale;
    *(float2*)&h[wave * FDIM + lane * 2] = o;
}

// ---------- edge cosine score (blended with distance prior): one wave per edge ----------
__global__ void k_edge_score(const float* __restrict__ h,
                             const int* __restrict__ i_idx, const int* __restrict__ j_idx,
                             const float* __restrict__ s_d, float* __restrict__ sval) {
    int wave = (blockIdx.x * blockDim.x + threadIdx.x) >> 5;
    int lane = threadIdx.x & 31;
    int i = i_idx[wave], j = j_idx[wave];
    float2 a = *(const float2*)&h[i * FDIM + lane * 2];
    float2 b = *(const float2*)&h[j * FDIM + lane * 2];
    float p = a.x * b.x + a.y * b.y;
    #pragma unroll
    for (int off = 16; off > 0; off >>= 1) p += __shfl_xor(p, off, 32);
    if (lane == 0)
        sval[wave] = OMEGA_C * fmaxf(p, 0.0f) + (1.0f - OMEGA_C) * s_d[wave];
}

// ---------- symmetrize via per-node block scan (replaces argsort+searchsorted) ----------
__global__ void k_symmetrize(const float* __restrict__ sval,
                             const int* __restrict__ i_idx, const int* __restrict__ j_idx,
                             float* __restrict__ vals_s, float* __restrict__ degS) {
    int e = blockIdx.x * blockDim.x + threadIdx.x;
    if (e >= EDG) return;
    int i = i_idx[e], j = j_idx[e];
    float v = sval[e];
    int rev = -1;
    int base = j * KNN;
    #pragma unroll
    for (int t = 0; t < KNN; ++t) {        // first match == stable-sort leftmost dup
        if (rev < 0 && j_idx[base + t] == i) rev = base + t;
    }
    float vs  = (rev >= 0) ? 0.5f * (v + sval[rev]) : v;
    float vs2 = (rev >= 0) ? 0.0f : vs;
    vals_s[e]       = vs;                  // forward copy
    vals_s[EDG + e] = vs2;                 // reverse copy (zero if reverse existed)
    atomAddF(&degS[i], vs);
    atomAddF(&degS[j], vs2);
}

__global__ void k_dinv_s(const float* __restrict__ deg, float* __restrict__ dinv, int n) {
    int t = blockIdx.x * blockDim.x + threadIdx.x;
    if (t < n) {
        float d = deg[t];
        dinv[t] = (d > 0.0f) ? rsqrtf(fmaxf(d, 1e-12f)) : 0.0f;
    }
}

__global__ void k_norm_vals(float* __restrict__ out, const float* __restrict__ vals_s,
                            const float* __restrict__ dinvS,
                            const int* __restrict__ i_idx, const int* __restrict__ j_idx) {
    int t = blockIdx.x * blockDim.x + threadIdx.x;   // 2E threads
    int r, c;
    if (t < EDG) { r = i_idx[t]; c = j_idx[t]; }
    else         { r = j_idx[t - EDG]; c = i_idx[t - EDG]; }
    out[t] = dinvS[r] * vals_s[t] * dinvS[c];
}

// ---------- host side ----------
extern "C" void kernel_launch(void* const* d_in, const int* in_sizes, int n_in,
                              void* d_out, int out_size, void* d_ws, size_t ws_size,
                              hipStream_t stream) {
    const float* features = (const float*)d_in[0];
    const float* W1       = (const float*)d_in[1];
    const float* b1       = (const float*)d_in[2];
    const float* W2       = (const float*)d_in[3];
    const float* b2       = (const float*)d_in[4];
    const float* s_d      = (const float*)d_in[5];
    const int*   adj_row  = (const int*)d_in[6];
    const int*   adj_col  = (const int*)d_in[7];
    const int*   i_idx    = (const int*)d_in[8];
    const int*   j_idx    = (const int*)d_in[9];

    char* ws = (char*)d_ws;
    size_t off = 0;
    auto alloc = [&](size_t bytes) -> void* {
        void* p = ws + off;
        off += (bytes + 255) & ~(size_t)255;
        return p;
    };
    float*          xw    = (float*)alloc((size_t)NN * FDIM * 4);
    float*          h     = (float*)alloc((size_t)NN * FDIM * 4);
    unsigned short* abf   = (unsigned short*)alloc((size_t)NN * FDIM * 2);
    unsigned short* wbt1  = (unsigned short*)alloc((size_t)FDIM * FDIM * 2);
    unsigned short* wbt2  = (unsigned short*)alloc((size_t)FDIM * FDIM * 2);
    float*          deg   = (float*)alloc((size_t)NN * 4);
    float*          dinv  = (float*)alloc((size_t)NN * 4);
    float*          degS  = (float*)alloc((size_t)NN * 4);
    float*          dinvS = (float*)alloc((size_t)NN * 4);
    float*          sval  = (float*)alloc((size_t)EDG * 4);

    float* out_norm = (float*)d_out;             // [2E]
    float* out_s    = (float*)d_out + 2 * EDG;   // [2E]

    const int B = 256;
    const int NF = NN * FDIM;                    // 2,560,000

    // bf16 conversions: features, and transpose-packed weights
    k_f2bf<<<NF / B, B, 0, stream>>>(features, abf, NF);
    k_w_t_bf<<<(FDIM * FDIM) / B, B, 0, stream>>>(W1, wbt1);
    k_w_t_bf<<<(FDIM * FDIM) / B, B, 0, stream>>>(W2, wbt2);

    // GCN degree / dinv (shared by both layers)
    k_fill<<<(NN + B - 1) / B, B, 0, stream>>>(deg, 1.0f, NN);
    k_deg_add<<<EADJ / B, B, 0, stream>>>(adj_col, deg, EADJ);
    k_rsqrt<<<(NN + B - 1) / B, B, 0, stream>>>(deg, dinv, NN);

    // Layer 1: WMMA GEMM -> self-loop init -> aggregate -> ReLU+bf16
    k_gemm_bf16<<<(NN / 16) * 4 / 8, B, 0, stream>>>(abf, wbt1, xw);   // 10000 waves
    k_selfloop_init<<<NF / B, B, 0, stream>>>(xw, dinv, b1, h);
    k_aggregate<<<(EADJ * FDIM) / B, B, 0, stream>>>(xw, adj_row, adj_col, dinv, h);
    k_relu_f2bf<<<NF / B, B, 0, stream>>>(h, abf, NF);

    // Layer 2: WMMA GEMM -> self-loop init -> aggregate -> row normalize
    k_gemm_bf16<<<(NN / 16) * 4 / 8, B, 0, stream>>>(abf, wbt2, xw);
    k_selfloop_init<<<NF / B, B, 0, stream>>>(xw, dinv, b2, h);
    k_aggregate<<<(EADJ * FDIM) / B, B, 0, stream>>>(xw, adj_row, adj_col, dinv, h);
    k_normalize<<<NN / 8, B, 0, stream>>>(h);                          // wave per node

    // Edge scores, symmetrization, symmetric normalization
    k_edge_score<<<EDG / 8, B, 0, stream>>>(h, i_idx, j_idx, s_d, sval); // wave per edge
    k_fill<<<(NN + B - 1) / B, B, 0, stream>>>(degS, 0.0f, NN);
    k_symmetrize<<<EDG / B, B, 0, stream>>>(sval, i_idx, j_idx, out_s, degS);
    k_dinv_s<<<(NN + B - 1) / B, B, 0, stream>>>(degS, dinvS, NN);
    k_norm_vals<<<(2 * EDG) / B, B, 0, stream>>>(out_norm, out_s, dinvS, i_idx, j_idx);
}
